// MultiHeadAttention_3023656976516
// MI455X (gfx1250) — matrix operand order repaired and checked by reading, not compiled, not run on hardware
//
#include <hip/hip_runtime.h>
#include <hip/hip_bf16.h>

#define DMODEL 1024
#define NHEADS 16
#define DK     64
#define BATCH  4
#define SEQ    2048

typedef _Float16 half_t;
typedef __attribute__((ext_vector_type(16))) _Float16 v16h;
typedef __attribute__((ext_vector_type(8)))  float    v8f;
typedef __attribute__((ext_vector_type(4)))  unsigned int u32x4;
typedef __attribute__((ext_vector_type(2)))  unsigned int u32x2;
typedef __attribute__((ext_vector_type(4)))  float    f32x4;

union FragA { v16h h; u32x4 u[2]; };
union FragC { v8f  f; float e[8]; };
union H8    { u32x4 u; half_t e[8]; };

// LDS row strides (halfs) chosen so 16 consecutive-row b128 gathers are
// bank-conflict-free while keeping 16-byte alignment:
//  32-col tiles: 56 halfs = 112 B  (28*r mod 64 banks: full permutation)
//  64-col tiles: 72 halfs = 144 B  (36*r mod 64 banks: full permutation)
#define PAD32 56
#define PAD64 72

// DPP8 selector for xor-by-m within 8-lane groups
constexpr unsigned dpp8_xor(unsigned m) {
  unsigned s = 0;
  for (unsigned i = 0; i < 8; ++i) s |= ((i ^ m) & 7u) << (3 * i);
  return s;
}

// max-reduce over the 16 lanes holding one matrix row:
// masks 1,2,4 on the VALU via DPP8; mask 8 via one ds_swizzle (SWAPX8).
__device__ __forceinline__ float rowmax16(float v) {
  v = fmaxf(v, __int_as_float(
                   __builtin_amdgcn_mov_dpp8(__float_as_int(v), dpp8_xor(1))));
  v = fmaxf(v, __int_as_float(
                   __builtin_amdgcn_mov_dpp8(__float_as_int(v), dpp8_xor(2))));
  v = fmaxf(v, __int_as_float(
                   __builtin_amdgcn_mov_dpp8(__float_as_int(v), dpp8_xor(4))));
  v = fmaxf(v, __int_as_float(
                   __builtin_amdgcn_ds_swizzle(__float_as_int(v), 0x201F)));
  return v;
}

// ---------------------------------------------------------------------------
// GEMM: Y[M,N] = (X[M,K] * W[N,K]^T + bias[N]) * out_scale  (M=8192, N=K=1024)
// Block tile 64x128, 8 waves (2x4), each wave 32x32 via 2x2 WMMA 16x16x32 f16.
// Software-pipelined: fetch tile k+1 into registers while WMMAs run on tile k;
// double-buffered LDS, one barrier per K-step.
// ---------------------------------------------------------------------------
template <bool IN_HALF, bool SCATTER>
__global__ __launch_bounds__(256) void gemm_x_wT(
    const void* __restrict__ Xv, const float* __restrict__ W,
    const float* __restrict__ bias, void* __restrict__ Yv, float out_scale) {
  constexpr int BM = 64, BN = 128, BK = 32;
  __shared__ __align__(16) half_t Xs[2][BM][PAD32];   // 14 KB
  __shared__ __align__(16) half_t Ws[2][BN][PAD32];   // 28 KB

  const int tid   = threadIdx.x;
  const int lane  = tid & 31;
  const int wave  = tid >> 5;
  const int group = lane >> 4;   // which 16-lane half
  const int lrow  = lane & 15;
  const int waveM = (wave & 1) << 5;   // 0 / 32
  const int waveN = (wave >> 1) << 5;  // 0 / 32 / 64 / 96
  const int m0 = blockIdx.x * BM;
  const int n0 = blockIdx.y * BN;

  const float*  Xf = (const float*)Xv;
  const half_t* Xh = (const half_t*)Xv;

  // per-thread staging registers (global -> reg -> LDS pipeline)
  f32x4 xs_f[2]; u32x2 xs_h[2]; f32x4 ws_f[4];

  auto fetch = [&](int k0) {
#pragma unroll
    for (int c = 0; c < 2; ++c) {            // X tile 64x32: 512 chunks of 4
      int idx = tid + (c << 8);
      int r = idx >> 3, kc = (idx & 7) << 2;
      if (IN_HALF)
        xs_h[c] = *(const u32x2*)(Xh + (size_t)(m0 + r) * DMODEL + k0 + kc);
      else
        xs_f[c] = *(const f32x4*)(Xf + (size_t)(m0 + r) * DMODEL + k0 + kc);
    }
#pragma unroll
    for (int c = 0; c < 4; ++c) {            // W tile 128x32: 1024 chunks of 4
      int idx = tid + (c << 8);
      int r = idx >> 3, kc = (idx & 7) << 2;
      ws_f[c] = *(const f32x4*)(W + (size_t)(n0 + r) * DMODEL + k0 + kc);
    }
  };
  auto stage = [&](int buf) {
#pragma unroll
    for (int c = 0; c < 2; ++c) {
      int idx = tid + (c << 8);
      int r = idx >> 3, kc = (idx & 7) << 2;
      if (IN_HALF) {
        *(u32x2*)&Xs[buf][r][kc] = xs_h[c];
      } else {
        Xs[buf][r][kc + 0] = (half_t)xs_f[c].x;
        Xs[buf][r][kc + 1] = (half_t)xs_f[c].y;
        Xs[buf][r][kc + 2] = (half_t)xs_f[c].z;
        Xs[buf][r][kc + 3] = (half_t)xs_f[c].w;
      }
    }
#pragma unroll
    for (int c = 0; c < 4; ++c) {
      int idx = tid + (c << 8);
      int r = idx >> 3, kc = (idx & 7) << 2;
      Ws[buf][r][kc + 0] = (half_t)ws_f[c].x;
      Ws[buf][r][kc + 1] = (half_t)ws_f[c].y;
      Ws[buf][r][kc + 2] = (half_t)ws_f[c].z;
      Ws[buf][r][kc + 3] = (half_t)ws_f[c].w;
    }
  };

  FragC acc[2][2];
#pragma unroll
  for (int mi = 0; mi < 2; ++mi)
#pragma unroll
    for (int ni = 0; ni < 2; ++ni)
#pragma unroll
      for (int r = 0; r < 8; ++r) acc[mi][ni].e[r] = 0.f;

  fetch(0);
  for (int kk = 0; kk < DMODEL / BK; ++kk) {
    const int buf = kk & 1;
    stage(buf);
    __syncthreads();
    if (kk + 1 < DMODEL / BK) fetch((kk + 1) * BK);  // overlap with WMMAs

    // --- gather fragments per documented 16-bit VGPR layouts --------------
    FragA a[2], b[2];
#pragma unroll
    for (int mi = 0; mi < 2; ++mi) {
      const half_t* base = &Xs[buf][waveM + mi * 16 + lrow][0];
      a[mi].u[0] = *(const u32x4*)(base + 8 * group);       // K 8g..8g+7
      a[mi].u[1] = *(const u32x4*)(base + 16 + 8 * group);  // K 16+8g..
    }
#pragma unroll
    for (int ni = 0; ni < 2; ++ni) {
      const half_t* base = &Ws[buf][waveN + ni * 16 + lrow][0];
      b[ni].u[0] = *(const u32x4*)(base + 8 * group);       // K 8g..8g+7
      b[ni].u[1] = *(const u32x4*)(base + 16 + 8 * group);
    }
#pragma unroll
    for (int mi = 0; mi < 2; ++mi)
#pragma unroll
      for (int ni = 0; ni < 2; ++ni)
        acc[mi][ni].f = __builtin_amdgcn_wmma_f32_16x16x32_f16(
            false, a[mi].h, false, b[ni].h, (short)0, acc[mi][ni].f, false,
            false);
  }

  // --- epilogue ----------------------------------------------------------
#pragma unroll
  for (int mi = 0; mi < 2; ++mi)
#pragma unroll
    for (int ni = 0; ni < 2; ++ni)
#pragma unroll
      for (int r = 0; r < 8; ++r) {
        int m = m0 + waveM + mi * 16 + (group << 3) + r;  // M = r + 8*group
        int n = n0 + waveN + ni * 16 + lrow;              // N = lane%16
        float val = (acc[mi][ni].e[r] + bias[n]) * out_scale;
        if (SCATTER) {
          int bb = m >> 11, t = m & (SEQ - 1);   // m = b*SEQ + t
          int hh = n >> 6, d = n & (DK - 1);     // n = h*DK + d
          ((half_t*)Yv)[(((size_t)bb * NHEADS + hh) * SEQ + t) * DK + d] =
              (half_t)val;
        } else {
          ((float*)Yv)[(size_t)m * DMODEL + n] = val;
        }
      }
}

// ---------------------------------------------------------------------------
// Flash attention: grid.y = b*H+h, grid.x = query block of 128.
// Each wave owns 16 queries; online softmax over 32-key tiles.
// Row sums on the matrix pipe (osum += P x Ones); running max reduced with
// DPP8 + one ds_swizzle. Double-buffered K/V tiles, fetch-ahead pipeline.
// ---------------------------------------------------------------------------
__global__ __launch_bounds__(256) void flash_attn(
    const half_t* __restrict__ q_ws, const half_t* __restrict__ k_ws,
    const half_t* __restrict__ v_ws, half_t* __restrict__ a_ws) {
  constexpr int QB = 128, KT = 32, NT = SEQ / KT;
  __shared__ __align__(16) half_t Qs[QB][PAD64];        // 18 KB
  __shared__ __align__(16) half_t Ks[2][KT][PAD64];     // 9 KB  (key-major)
  __shared__ __align__(16) half_t Vt[2][DK][PAD32];     // 14 KB (transposed)
  __shared__ __align__(16) half_t Ps[8][16][PAD32];     // 14 KB (per-wave P)

  const int tid = threadIdx.x;
  const int lane = tid & 31, wave = tid >> 5;
  const int group = lane >> 4, lrow = lane & 15;
  const int bh = blockIdx.y;              // b*H + h
  const int b = bh >> 4, h = bh & 15;
  const int q0 = blockIdx.x * QB;

  const size_t head_off = (size_t)bh * SEQ * DK;
  const half_t* Qg = q_ws + head_off + (size_t)q0 * DK;
  const half_t* Kg = k_ws + head_off;
  const half_t* Vg = v_ws + head_off;

  // --- load Q block 128x64 to LDS ---------------------------------------
#pragma unroll
  for (int c = 0; c < 4; ++c) {
    int idx = tid + (c << 8);  // 1024 chunks of 8 halfs
    int r = idx >> 3, d = (idx & 7) << 3;
    *(u32x4*)&Qs[r][d] = *(const u32x4*)(Qg + (size_t)r * DK + d);
  }
  __syncthreads();

  // Q A-fragments for the two 32-wide K-steps of DK=64 (kept in registers)
  FragA qf[2];
#pragma unroll
  for (int s = 0; s < 2; ++s) {
    const half_t* base = &Qs[wave * 16 + lrow][s * 32];
    qf[s].u[0] = *(const u32x4*)(base + 8 * group);
    qf[s].u[1] = *(const u32x4*)(base + 16 + 8 * group);
  }

  // all-ones B fragment: D[m][n] = sum_k P[m][k] for every n (row-sum bcast)
  FragA onesf;
#pragma unroll
  for (int i = 0; i < 16; ++i) onesf.h[i] = (_Float16)1.0f;

  FragC o[4], osum;
  float row_max[8];
#pragma unroll
  for (int r = 0; r < 8; ++r) { row_max[r] = -1e30f; osum.e[r] = 0.f; }
#pragma unroll
  for (int ns = 0; ns < 4; ++ns)
#pragma unroll
    for (int r = 0; r < 8; ++r) o[ns].e[r] = 0.f;

  // fetch-ahead staging registers for K/V tiles
  const int tr = tid >> 3, tdc = (tid & 7) << 3;
  H8 kreg, vreg;
  auto fetch = [&](int kt) {
    kreg.u = *(const u32x4*)(Kg + (size_t)(kt + tr) * DK + tdc);
    vreg.u = *(const u32x4*)(Vg + (size_t)(kt + tr) * DK + tdc);
  };
  auto stage = [&](int buf) {
    *(u32x4*)&Ks[buf][tr][tdc] = kreg.u;
#pragma unroll
    for (int j = 0; j < 8; ++j) Vt[buf][tdc + j][tr] = vreg.e[j];  // transpose
  };

  fetch(0);
  for (int t = 0; t < NT; ++t) {
    const int buf = t & 1;
    stage(buf);
    __syncthreads();
    if (t + 1 < NT) fetch((t + 1) * KT);   // overlap with compute below

    // --- S = Q * K^T : two 16-key column blocks, 2 K-steps each ----------
    FragC Sf[2];
#pragma unroll
    for (int j = 0; j < 2; ++j) {
      v8f c = {};
#pragma unroll
      for (int s = 0; s < 2; ++s) {
        FragA bb;
        const half_t* base = &Ks[buf][j * 16 + lrow][s * 32 + 8 * group];
        bb.u[0] = *(const u32x4*)(base);
        bb.u[1] = *(const u32x4*)(base + 16);
        c = __builtin_amdgcn_wmma_f32_16x16x32_f16(false, qf[s].h, false,
                                                   bb.h, (short)0, c, false,
                                                   false);
      }
      Sf[j].f = c;
    }

    // --- online softmax max update (row = r + 8*group, over 16 lanes) ----
    float alpha[8];
#pragma unroll
    for (int r = 0; r < 8; ++r) {
      float s0 = Sf[0].e[r], s1 = Sf[1].e[r];  // scale folded into Q proj
      float mloc = rowmax16(fmaxf(s0, s1));
      float mnew = fmaxf(row_max[r], mloc);
      alpha[r] = __expf(row_max[r] - mnew);
      row_max[r] = mnew;
      float p0 = __expf(s0 - mnew), p1 = __expf(s1 - mnew);
      Ps[wave][(group << 3) + r][lrow]      = (half_t)p0;
      Ps[wave][(group << 3) + r][16 + lrow] = (half_t)p1;
    }
#pragma unroll
    for (int ns = 0; ns < 4; ++ns)
#pragma unroll
      for (int r = 0; r < 8; ++r) o[ns].e[r] *= alpha[r];
#pragma unroll
    for (int r = 0; r < 8; ++r) osum.e[r] *= alpha[r];

    // --- O += P*V and osum += P*Ones (P via LDS to reach A-layout) -------
    FragA pf;
    const half_t* pbase = &Ps[wave][lrow][0];
    pf.u[0] = *(const u32x4*)(pbase + 8 * group);
    pf.u[1] = *(const u32x4*)(pbase + 16 + 8 * group);
#pragma unroll
    for (int ns = 0; ns < 4; ++ns) {
      FragA bb;
      const half_t* vbase = &Vt[buf][ns * 16 + lrow][8 * group];
      bb.u[0] = *(const u32x4*)(vbase);
      bb.u[1] = *(const u32x4*)(vbase + 16);
      o[ns].f = __builtin_amdgcn_wmma_f32_16x16x32_f16(
          false, pf.h, false, bb.h, (short)0, o[ns].f, false, false);
    }
    osum.f = __builtin_amdgcn_wmma_f32_16x16x32_f16(
        false, pf.h, false, onesf.h, (short)0, osum.f, false, false);
  }

  // --- normalize and store f16 into [B, T, H*DK] -------------------------
  float inv[8];
#pragma unroll
  for (int r = 0; r < 8; ++r) inv[r] = 1.0f / osum.e[r];
  half_t* Og = a_ws + ((size_t)b * SEQ + q0 + wave * 16) * DMODEL + h * DK;
#pragma unroll
  for (int ns = 0; ns < 4; ++ns)
#pragma unroll
    for (int r = 0; r < 8; ++r) {
      int t = (group << 3) + r;
      int d = ns * 16 + lrow;
      Og[(size_t)t * DMODEL + d] = (half_t)(o[ns].e[r] * inv[r]);
    }
}

// ---------------------------------------------------------------------------
extern "C" void kernel_launch(void* const* d_in, const int* in_sizes, int n_in,
                              void* d_out, int out_size, void* d_ws,
                              size_t ws_size, hipStream_t stream) {
  const float* query = (const float*)d_in[0];
  const float* key   = (const float*)d_in[1];
  const float* value = (const float*)d_in[2];
  const float* Wq = (const float*)d_in[3];
  const float* bq = (const float*)d_in[4];
  const float* Wk = (const float*)d_in[5];
  const float* bk = (const float*)d_in[6];
  const float* Wv = (const float*)d_in[7];
  const float* bv = (const float*)d_in[8];
  const float* Wo = (const float*)d_in[9];
  const float* bo = (const float*)d_in[10];

  const size_t QKV = (size_t)BATCH * NHEADS * SEQ * DK;  // 8,388,608 halfs
  half_t* q_ws = (half_t*)d_ws;
  half_t* k_ws = q_ws + QKV;
  half_t* v_ws = k_ws + QKV;
  half_t* a_ws = v_ws + QKV;  // [B*T, DMODEL] f16; total ws = 64 MB (fits L2)

  const float qscale = 0.125f;  // 1/sqrt(DK), folded into Q projection

  dim3 gp((BATCH * SEQ) / 64, DMODEL / 128);  // 128 x 8 blocks
  gemm_x_wT<false, true><<<gp, 256, 0, stream>>>(query, Wq, bq, q_ws, qscale);
  gemm_x_wT<false, true><<<gp, 256, 0, stream>>>(key,   Wk, bk, k_ws, 1.0f);
  gemm_x_wT<false, true><<<gp, 256, 0, stream>>>(value, Wv, bv, v_ws, 1.0f);

  dim3 ga(SEQ / 128, BATCH * NHEADS);  // 16 x 64 blocks
  flash_attn<<<ga, 256, 0, stream>>>(q_ws, k_ws, v_ws, a_ws);

  gemm_x_wT<true, false><<<gp, 256, 0, stream>>>(a_ws, Wo, bo, d_out, 1.0f);
}
